// SelectiveMLP_80994493268149
// MI455X (gfx1250) — matrix operand order repaired and checked by reading, not compiled, not run on hardware
//
#include <hip/hip_runtime.h>

// ---------------------------------------------------------------------------
// SelectiveMLP for MI455X (gfx1250, wave32, WMMA bf16, async global->LDS)
// out = relu(x @ fc1_w[idx].T + bias1[idx]) @ fc2_w_t[idx] + bias2
// 2 x (4096^3) GEMMs = 275 GFLOP -> matrix-unit bound; bf16 WMMA, f32 accum.
// ---------------------------------------------------------------------------

typedef __attribute__((ext_vector_type(16))) __bf16 v16bf;
typedef __attribute__((ext_vector_type(8)))  float  v8f;

#define DIM 4096   // D_IN == D_OUT == N_TOK == K_SEL == 4096

__device__ __forceinline__ unsigned short f2bf(float f) {
    unsigned int u = __float_as_uint(f);
    u += 0x7FFFu + ((u >> 16) & 1u);          // round-to-nearest-even
    return (unsigned short)(u >> 16);
}

// 16B global -> LDS async copy (GLOBAL_LOAD_ASYNC_TO_LDS_B128, ASYNCcnt).
// Operand order per VGLOBAL encoding: VDST = LDS byte address, VADDR = 64-bit
// global address, SADDR = off. LDS offset = low 32 bits of flat shared addr.
__device__ __forceinline__ void async_ld_b128(unsigned lds_off, const void* g) {
    asm volatile("global_load_async_to_lds_b128 %0, %1, off"
                 :: "v"(lds_off), "v"(g) : "memory");
}
__device__ __forceinline__ void wait_async0() {
    asm volatile("s_wait_asynccnt 0x0" ::: "memory");
}

// ---- fp32 -> bf16 bulk convert (x) ----------------------------------------
__global__ void k_convert_bf16(const float* __restrict__ src,
                               unsigned short* __restrict__ dst) {
    int i = blockIdx.x * blockDim.x + threadIdx.x;          // one float4 each
    float4 f = reinterpret_cast<const float4*>(src)[i];
    uint2 p;
    p.x = (unsigned)f2bf(f.x) | ((unsigned)f2bf(f.y) << 16);
    p.y = (unsigned)f2bf(f.z) | ((unsigned)f2bf(f.w) << 16);
    reinterpret_cast<uint2*>(dst)[i] = p;
}

// ---- gather fc1 rows (and bias1) -> bf16 ----------------------------------
__global__ void k_gather_w1(const float* __restrict__ fc1,
                            const float* __restrict__ bias1,
                            const int* __restrict__ idx,
                            unsigned short* __restrict__ w1b,
                            float* __restrict__ b1g) {
    int r   = blockIdx.y;
    int row = idx[r];
    const float*    src = fc1 + (size_t)row * DIM;
    unsigned short* dst = w1b + (size_t)r   * DIM;
    int base = blockIdx.x * 1024 + threadIdx.x * 4;
    float4 f = *reinterpret_cast<const float4*>(src + base);
    uint2 p;
    p.x = (unsigned)f2bf(f.x) | ((unsigned)f2bf(f.y) << 16);
    p.y = (unsigned)f2bf(f.z) | ((unsigned)f2bf(f.w) << 16);
    *reinterpret_cast<uint2*>(dst + base) = p;
    if (blockIdx.x == 0 && threadIdx.x == 0) b1g[r] = bias1[row];
}

// ---- gather + transpose fc2_w_t rows -> w2bT[D_OUT][K_SEL] bf16 -----------
__global__ void k_gather_w2t(const float* __restrict__ fc2t,
                             const int* __restrict__ idx,
                             unsigned short* __restrict__ w2bT) {
    __shared__ float tile[32][33];
    int k0 = blockIdx.x * 32, o0 = blockIdx.y * 32;
    int tx = threadIdx.x, ty = threadIdx.y;
#pragma unroll
    for (int i = 0; i < 4; ++i) {
        int k = k0 + ty + i * 8;
        tile[ty + i * 8][tx] = fc2t[(size_t)idx[k] * DIM + o0 + tx];  // coalesced read
    }
    __syncthreads();
#pragma unroll
    for (int i = 0; i < 4; ++i) {
        int o = o0 + ty + i * 8;
        w2bT[(size_t)o * DIM + k0 + tx] = f2bf(tile[tx][ty + i * 8]); // coalesced write
    }
}

// ---- NT GEMM: C[M,N] = act(A[M,:]·B[N,:] + bias[N]) -----------------------
// A, B row-major bf16, reduction dim contiguous (ld = 4096).
// Block tile 128x128, 256 threads = 8 waves (2 M x 4 N), wave tile 64x32 =
// 4x2 v_wmma_f32_16x16x32_bf16. Double-buffered LDS fed by async-to-LDS.
template <bool RELU, bool OUT_BF16>
__global__ __launch_bounds__(256) void k_gemm_nt(
    const unsigned short* __restrict__ A,
    const unsigned short* __restrict__ B,
    const float* __restrict__ bias,
    void* __restrict__ Cout) {
    constexpr int BM = 128, BN = 128, KS = 32;
    constexpr int LST = 40;            // padded LDS row stride (elements)
    constexpr int BUFE = BM * LST;     // elements per buffer (5120)
    __shared__ __align__(16) unsigned short smA[2 * BUFE];  // 20 KB
    __shared__ __align__(16) unsigned short smB[2 * BUFE];  // 20 KB

    const int tid   = threadIdx.x;
    const int lane  = tid & 31;
    const int wave  = tid >> 5;
    const int waveM = (wave & 1) * 64;   // 0 / 64
    const int waveN = (wave >> 1) * 32;  // 0 / 32 / 64 / 96
    const int blkM  = blockIdx.y * BM;
    const int blkN  = blockIdx.x * BN;

    // staging: each thread owns two 16B chunks of A and two of B per K-step
    const int r0 = tid >> 2;             // rows 0..63
    const int r1 = r0 + 64;              // rows 64..127
    const int c0 = (tid & 3) * 8;        // element column 0/8/16/24
    const unsigned short* gA0 = A + (size_t)(blkM + r0) * DIM + c0;
    const unsigned short* gA1 = A + (size_t)(blkM + r1) * DIM + c0;
    const unsigned short* gB0 = B + (size_t)(blkN + r0) * DIM + c0;
    const unsigned short* gB1 = B + (size_t)(blkN + r1) * DIM + c0;

    const unsigned ldsA  = (unsigned)(size_t)smA;            // LDS byte offset
    const unsigned ldsB  = (unsigned)(size_t)smB;
    const unsigned off0  = (unsigned)(r0 * LST + c0) * 2u;
    const unsigned off1  = (unsigned)(r1 * LST + c0) * 2u;
    const unsigned bufB  = (unsigned)BUFE * 2u;              // buffer stride bytes

    auto stage = [&](int k0, int b) {
        unsigned bo = (unsigned)b * bufB;
        async_ld_b128(ldsA + bo + off0, gA0 + k0);
        async_ld_b128(ldsA + bo + off1, gA1 + k0);
        async_ld_b128(ldsB + bo + off0, gB0 + k0);
        async_ld_b128(ldsB + bo + off1, gB1 + k0);
    };

    const v8f vzero = {0.f, 0.f, 0.f, 0.f, 0.f, 0.f, 0.f, 0.f};
    v8f acc[4][2];
#pragma unroll
    for (int mi = 0; mi < 4; ++mi)
#pragma unroll
        for (int ni = 0; ni < 2; ++ni) acc[mi][ni] = vzero;

    const int fr   = lane & 15;   // M (A) / N (B) index inside 16-wide tile
    const int half = lane >> 4;   // which K-half this lane carries
    const unsigned aE0 = half * 8,  aE1 = 16 + half * 8;   // A: K 0-7/8-15, 16-23/24-31
    const unsigned bE0 = half * 8,  bE1 = 16 + half * 8;   // B: K 0-7/8-15, 16-23/24-31

    constexpr int NK = DIM / KS;  // 128
    stage(0, 0);
    wait_async0();
    __syncthreads();

    for (int k = 0; k < NK; ++k) {
        if (k + 1 < NK) stage((k + 1) * KS, (k + 1) & 1);   // prefetch next slab

        const unsigned short* bA = smA + (k & 1) * BUFE;
        const unsigned short* bB = smB + (k & 1) * BUFE;

        v16bf af[4], bfr[2];
#pragma unroll
        for (int mi = 0; mi < 4; ++mi) {
            const unsigned short* base = bA + (waveM + mi * 16 + fr) * LST;
            union { uint4 u[2]; v16bf v; } f;
            f.u[0] = *reinterpret_cast<const uint4*>(base + aE0);
            f.u[1] = *reinterpret_cast<const uint4*>(base + aE1);
            af[mi] = f.v;
        }
#pragma unroll
        for (int ni = 0; ni < 2; ++ni) {
            const unsigned short* base = bB + (waveN + ni * 16 + fr) * LST;
            union { uint4 u[2]; v16bf v; } f;
            f.u[0] = *reinterpret_cast<const uint4*>(base + bE0);
            f.u[1] = *reinterpret_cast<const uint4*>(base + bE1);
            bfr[ni] = f.v;
        }
#pragma unroll
        for (int ni = 0; ni < 2; ++ni)
#pragma unroll
            for (int mi = 0; mi < 4; ++mi)
                acc[mi][ni] = __builtin_amdgcn_wmma_f32_16x16x32_bf16(
                    false, af[mi], false, bfr[ni], (short)0, acc[mi][ni], false, false);

        wait_async0();     // next slab landed in LDS
        __syncthreads();   // everyone done reading current slab
    }

    // Epilogue. C/D layout: VGPR r, lane l -> M = r + 8*(l>>4), N = l&15.
    const int cN = lane & 15;
    const int cM = (lane >> 4) * 8;
#pragma unroll
    for (int mi = 0; mi < 4; ++mi)
#pragma unroll
        for (int ni = 0; ni < 2; ++ni) {
            int gn = blkN + waveN + ni * 16 + cN;
            float bv = bias[gn];
#pragma unroll
            for (int r = 0; r < 8; ++r) {
                int gm = blkM + waveM + mi * 16 + cM + r;
                float v = acc[mi][ni][r] + bv;
                if (RELU) v = fmaxf(v, 0.0f);
                if (OUT_BF16)
                    reinterpret_cast<unsigned short*>(Cout)[(size_t)gm * DIM + gn] = f2bf(v);
                else
                    reinterpret_cast<float*>(Cout)[(size_t)gm * DIM + gn] = v;
            }
        }
}

// ---------------------------------------------------------------------------
extern "C" void kernel_launch(void* const* d_in, const int* in_sizes, int n_in,
                              void* d_out, int out_size, void* d_ws, size_t ws_size,
                              hipStream_t stream) {
    const float* x     = (const float*)d_in[0];
    const float* fc1   = (const float*)d_in[1];
    const float* fc2t  = (const float*)d_in[2];
    const float* bias1 = (const float*)d_in[3];
    const float* bias2 = (const float*)d_in[4];
    const int*   idx   = (const int*)d_in[5];

    const size_t MAT = (size_t)DIM * DIM;          // 16M elements
    unsigned short* xb   = (unsigned short*)d_ws;  // 32 MB each
    unsigned short* w1b  = xb   + MAT;
    unsigned short* w2bT = w1b  + MAT;
    unsigned short* hbuf = w2bT + MAT;
    float*          b1g  = (float*)(hbuf + MAT);   // 16 KB

    // Stage 0: convert + gather (memory-bound, ~15us of HBM traffic total)
    k_convert_bf16<<<dim3((unsigned)(MAT / 1024)), 256, 0, stream>>>(x, xb);
    k_gather_w1<<<dim3(4, DIM), 256, 0, stream>>>(fc1, bias1, idx, w1b, b1g);
    k_gather_w2t<<<dim3(DIM / 32, DIM / 32), dim3(32, 8), 0, stream>>>(fc2t, idx, w2bT);

    // Stage 1: h = relu(x @ w1^T + b1)  (bf16 out for the second GEMM)
    dim3 gemmGrid(DIM / 128, DIM / 128);
    k_gemm_nt<true, true><<<gemmGrid, 256, 0, stream>>>(xb, w1b, b1g, hbuf);
    // Stage 2: out = h @ w2 + b2  (fp32 out)
    k_gemm_nt<false, false><<<gemmGrid, 256, 0, stream>>>(hbuf, w2bT, bias2, d_out);
}
